// LocalAttention_56736517980256
// MI455X (gfx1250) — matrix operand order repaired and checked by reading, not compiled, not run on hardware
//
#include <hip/hip_runtime.h>
#include <hip/hip_bf16.h>

typedef __attribute__((ext_vector_type(16))) __bf16 v16bf;
typedef __attribute__((ext_vector_type(8)))  float  v8f;
typedef unsigned int u32;
typedef __attribute__((ext_vector_type(4))) u32 u32x4;
typedef __attribute__((ext_vector_type(8))) int i32x8;
typedef __attribute__((ext_vector_type(4))) int i32x4;

#define BATCH 4
#define HH    64
#define WW    64
#define CC    128
#define NH    4
#define HD    32
#define KK    7
#define NPIX  (BATCH * HH * WW)          // 16384
#define C3    (3 * CC)                   // 384
#define RPBN  (NH * (2*KK-1) * (2*KK-1)) // 676

// ---------------------------------------------------------------------------
// helpers
// ---------------------------------------------------------------------------
__device__ __forceinline__ void bf2x(unsigned int wd, float& lo, float& hi) {
    union { unsigned int i; float f; } a, b;
    a.i = wd << 16;
    b.i = wd & 0xffff0000u;
    lo = a.f; hi = b.f;
}

// Stage a (krows x 16) f32 tile of `w` (row length `rowlen` elems, tile starts
// at column n0) into LDS offset 0 via the Tensor Data Mover.
// D# layout per CDNA5 ISA:
//   g0: [1:0]=count=1, [63:32]=lds_addr(0), [120:64]=global_addr, [127:126]=type=2
//   g1: [17:16]=data_size=2 (4B), [79:48]=tensor_dim0, [111:80]=tensor_dim1,
//       [127:112]=tile_dim0=16, [143:128]=tile_dim1=krows, [207:160]=dim0_stride
// Toolchain here exposes the 6-arg builtin:
//   (uint32x4 g0, int32x8 g1, int32x4 g2, int32x4 g3, int32x8 pad, i32 cpol)
__device__ __forceinline__ void tdm_stage_tile(const float* w, int n0,
                                               int rowlen, int krows) {
    unsigned long long ga = (unsigned long long)(const void*)w
                          + (unsigned long long)n0 * 4ull;
    u32x4 g0 = { 1u,                                   // count = 1 valid D#
                 0u,                                   // lds_addr = 0 (dyn LDS)
                 (u32)(ga & 0xffffffffull),
                 (u32)((ga >> 32) & 0x01ffffffu) | 0x80000000u };  // type=2
    i32x8 g1 = { (int)(2u << 16),                      // data_size = 4 bytes
                 (int)((rowlen & 0xffff) << 16),       // tensor_dim0 lo16
                 (int)(((rowlen >> 16) & 0xffff) | ((krows & 0xffff) << 16)),
                 (int)(((krows  >> 16) & 0xffff) | (16 << 16)),   // tile_dim0=16
                 (int)(krows & 0xffff),                // tile_dim1 = krows
                 rowlen,                               // tensor_dim0_stride lo32
                 0, 0 };
    i32x4 gz4 = { 0, 0, 0, 0 };
    i32x8 gz8 = { 0, 0, 0, 0, 0, 0, 0, 0 };
    __builtin_amdgcn_tensor_load_to_lds(g0, g1, gz4, gz4, gz8, 0);
}

// ---------------------------------------------------------------------------
// Kernel 1: qkv = x @ w_qkv + b_qkv   (f32 in, bf16 out, WMMA bf16)
// Block = 8 waves sharing one N tile; B tile (128x16 f32) staged in LDS by TDM.
// Each wave computes one 16x16 output tile, K=128 in 4 WMMA steps.
// ---------------------------------------------------------------------------
__global__ void __launch_bounds__(256) qkv_gemm_kernel(
    const float* __restrict__ x, const float* __restrict__ w,
    const float* __restrict__ bias,
    __bf16* __restrict__ qarr, __bf16* __restrict__ karr, __bf16* __restrict__ varr)
{
    extern __shared__ float bstage[];                 // [128][16] f32, 8 KB
    const int TN    = C3 / 16;                        // 24
    const int tn    = blockIdx.x % TN;
    const int tmblk = blockIdx.x / TN;
    const int waveid = threadIdx.x >> 5;
    const int lane   = threadIdx.x & 31;
    const int m0     = (tmblk * 8 + waveid) * 16;
    const int n0     = tn * 16;

    // wave 0 DMAs the whole 128x16 weight tile into LDS
    if (threadIdx.x < 32) {
        tdm_stage_tile(w, n0, C3, CC);
        __builtin_amdgcn_s_wait_tensorcnt(0);
    }
    __syncthreads();

    const int row = lane & 15;
    const int col = lane & 15;
    const int akb = (lane < 16) ? 0 : 16;             // A K-base per half-wave
    const int bkb = (lane < 16) ? 0 : 16;             // B K-base per half-wave

    v8f c = {};
    for (int k0 = 0; k0 < CC; k0 += 32) {
        v16bf a, bm;
        const float* ap = x + (size_t)(m0 + row) * CC + k0;
        #pragma unroll
        for (int r = 0; r < 4; ++r) {
            int ka = akb + 2 * r;
            a[2*r]     = (__bf16)ap[ka];
            a[2*r + 1] = (__bf16)ap[ka + 1];
            int kb = akb + 8 + 2 * r;
            a[8 + 2*r]     = (__bf16)ap[kb];
            a[8 + 2*r + 1] = (__bf16)ap[kb + 1];
        }
        const float* bp = bstage + (size_t)k0 * 16 + col;
        #pragma unroll
        for (int r = 0; r < 8; ++r) {
            int kk = bkb + 2 * r;
            bm[2*r]     = (__bf16)bp[kk * 16];
            bm[2*r + 1] = (__bf16)bp[(kk + 1) * 16];
        }
        c = __builtin_amdgcn_wmma_f32_16x16x32_bf16(
                false, a, false, bm, (short)0, c, false, false);
    }

    // epilogue: bias + scale + split into q/k/v bf16 arrays [pix][C]
    const int   n   = n0 + col;
    const float bv  = bias[n];
    const bool  isq = (n < CC);
    const float scl = isq ? 0.17677669529663687f : 1.0f;  // 1/sqrt(32) for q
    __bf16* dst = isq ? qarr : ((n < 2 * CC) ? karr : varr);
    const int nn = n & (CC - 1);
    #pragma unroll
    for (int r = 0; r < 8; ++r) {
        int mrow = m0 + ((lane < 16) ? r : (r + 8));
        float val = (c[r] + bv) * scl;
        dst[(size_t)mrow * CC + nn] = (__bf16)val;
    }
}

// ---------------------------------------------------------------------------
// Kernel 2: neighborhood attention core, flash-style online softmax.
// One thread per (pixel, head); 49 neighbors x 32 dims. RPB staged in LDS.
// ---------------------------------------------------------------------------
__global__ void __launch_bounds__(256) attn_kernel(
    const __bf16* __restrict__ qarr, const __bf16* __restrict__ karr,
    const __bf16* __restrict__ varr, const float* __restrict__ rpb,
    __bf16* __restrict__ oarr)
{
    __shared__ float rpb_s[RPBN];
    for (int t = threadIdx.x; t < RPBN; t += 256) rpb_s[t] = rpb[t];
    __syncthreads();

    const int t    = blockIdx.x * 256 + threadIdx.x;
    const int head = t & (NH - 1);
    const int pix  = t >> 2;
    const int b    = pix >> 12;
    const int ij   = pix & 4095;
    const int i    = ij >> 6;
    const int j    = ij & 63;
    const int h0   = min(max(i - KK / 2, 0), HH - KK);
    const int w0   = min(max(j - KK / 2, 0), WW - KK);

    float q[HD];
    {
        const uint4* qp = (const uint4*)(qarr + (size_t)pix * CC + head * HD);
        #pragma unroll
        for (int p = 0; p < 4; ++p) {
            uint4 u = qp[p];
            bf2x(u.x, q[p*8+0], q[p*8+1]);
            bf2x(u.y, q[p*8+2], q[p*8+3]);
            bf2x(u.z, q[p*8+4], q[p*8+5]);
            bf2x(u.w, q[p*8+6], q[p*8+7]);
        }
    }

    float m = -INFINITY, l = 0.0f;
    float acc[HD];
    #pragma unroll
    for (int d = 0; d < HD; ++d) acc[d] = 0.0f;

    const float* rb = rpb_s + head * (2*KK-1) * (2*KK-1);

    for (int ii = 0; ii < KK; ++ii) {
        const int ki = h0 + ii;
        const int bh = ki - i + (KK - 1);
        const size_t rowbase = ((size_t)(b * 4096 + ki * 64 + w0)) * CC + head * HD;
        const __bf16* krow = karr + rowbase;
        const __bf16* vrow = varr + rowbase;
        if (ii + 1 < KK) {  // prefetch next key row -> global_prefetch_b8
            __builtin_prefetch(karr + rowbase + (size_t)64 * CC, 0, 3);
            __builtin_prefetch(varr + rowbase + (size_t)64 * CC, 0, 3);
        }
        for (int jj = 0; jj < KK; ++jj) {
            const int bw = (w0 + jj) - j + (KK - 1);
            float s = 0.0f;
            {
                const uint4* kp = (const uint4*)(krow + jj * CC);
                #pragma unroll
                for (int p = 0; p < 4; ++p) {
                    uint4 u = kp[p];
                    float kv[8];
                    bf2x(u.x, kv[0], kv[1]);
                    bf2x(u.y, kv[2], kv[3]);
                    bf2x(u.z, kv[4], kv[5]);
                    bf2x(u.w, kv[6], kv[7]);
                    #pragma unroll
                    for (int e = 0; e < 8; ++e) s = fmaf(q[p*8+e], kv[e], s);
                }
            }
            s += rb[bh * (2*KK-1) + bw];

            const float mn   = fmaxf(m, s);
            const float corr = __expf(m - mn);
            const float prob = __expf(s - mn);
            l = l * corr + prob;
            m = mn;
            #pragma unroll
            for (int d = 0; d < HD; ++d) acc[d] *= corr;
            {
                const uint4* vp = (const uint4*)(vrow + jj * CC);
                #pragma unroll
                for (int p = 0; p < 4; ++p) {
                    uint4 u = vp[p];
                    float vv[8];
                    bf2x(u.x, vv[0], vv[1]);
                    bf2x(u.y, vv[2], vv[3]);
                    bf2x(u.z, vv[4], vv[5]);
                    bf2x(u.w, vv[6], vv[7]);
                    #pragma unroll
                    for (int e = 0; e < 8; ++e) acc[p*8+e] = fmaf(prob, vv[e], acc[p*8+e]);
                }
            }
        }
    }

    const float inv = 1.0f / l;
    __bf16* op = oarr + (size_t)pix * CC + head * HD;
    #pragma unroll
    for (int d = 0; d < HD; ++d) op[d] = (__bf16)(acc[d] * inv);
}

// ---------------------------------------------------------------------------
// Kernel 3: out = attn_out @ w_proj + b_proj   (bf16 A, f32 W -> f32 out)
// Same block shape as K1: 8 waves share one TDM-staged 128x16 weight tile.
// ---------------------------------------------------------------------------
__global__ void __launch_bounds__(256) proj_gemm_kernel(
    const __bf16* __restrict__ aarr, const float* __restrict__ w,
    const float* __restrict__ bias, float* __restrict__ out)
{
    extern __shared__ float bstage[];                 // [128][16] f32, 8 KB
    const int TN    = CC / 16;                        // 8
    const int tn    = blockIdx.x % TN;
    const int tmblk = blockIdx.x / TN;
    const int waveid = threadIdx.x >> 5;
    const int lane   = threadIdx.x & 31;
    const int m0     = (tmblk * 8 + waveid) * 16;
    const int n0     = tn * 16;

    if (threadIdx.x < 32) {
        tdm_stage_tile(w, n0, CC, CC);
        __builtin_amdgcn_s_wait_tensorcnt(0);
    }
    __syncthreads();

    const int row = lane & 15;
    const int col = lane & 15;
    const int akb = (lane < 16) ? 0 : 16;
    const int bkb = (lane < 16) ? 0 : 16;

    v8f c = {};
    for (int k0 = 0; k0 < CC; k0 += 32) {
        v16bf a, bm;
        const __bf16* ap = aarr + (size_t)(m0 + row) * CC + k0;
        #pragma unroll
        for (int r = 0; r < 4; ++r) {
            int ka = akb + 2 * r;
            a[2*r]     = ap[ka];
            a[2*r + 1] = ap[ka + 1];
            int kb = akb + 8 + 2 * r;
            a[8 + 2*r]     = ap[kb];
            a[8 + 2*r + 1] = ap[kb + 1];
        }
        const float* bp = bstage + (size_t)k0 * 16 + col;
        #pragma unroll
        for (int r = 0; r < 8; ++r) {
            int kk = bkb + 2 * r;
            bm[2*r]     = (__bf16)bp[kk * 16];
            bm[2*r + 1] = (__bf16)bp[(kk + 1) * 16];
        }
        c = __builtin_amdgcn_wmma_f32_16x16x32_bf16(
                false, a, false, bm, (short)0, c, false, false);
    }

    const int   n  = n0 + col;
    const float bv = bias[n];
    #pragma unroll
    for (int r = 0; r < 8; ++r) {
        int mrow = m0 + ((lane < 16) ? r : (r + 8));
        out[(size_t)mrow * CC + n] = c[r] + bv;
    }
}

// ---------------------------------------------------------------------------
// launch
// ---------------------------------------------------------------------------
extern "C" void kernel_launch(void* const* d_in, const int* in_sizes, int n_in,
                              void* d_out, int out_size, void* d_ws, size_t ws_size,
                              hipStream_t stream)
{
    const float* x      = (const float*)d_in[0];
    const float* w_qkv  = (const float*)d_in[1];
    const float* b_qkv  = (const float*)d_in[2];
    const float* rpb    = (const float*)d_in[3];
    const float* w_proj = (const float*)d_in[4];
    const float* b_proj = (const float*)d_in[5];
    float*       out    = (float*)d_out;

    char* wsb = (char*)d_ws;
    const size_t arr_bytes = (size_t)NPIX * CC * sizeof(unsigned short); // 4 MB
    __bf16* qarr = (__bf16*)(wsb + 0 * arr_bytes);
    __bf16* karr = (__bf16*)(wsb + 1 * arr_bytes);
    __bf16* varr = (__bf16*)(wsb + 2 * arr_bytes);
    __bf16* oarr = (__bf16*)(wsb + 3 * arr_bytes);

    const size_t stage_bytes = (size_t)CC * 16 * sizeof(float); // 8 KB LDS

    // K1: 24 N-tiles x 128 M-blocks (8 waves/block, 1 M-tile per wave)
    {
        int blocks = (C3 / 16) * (NPIX / 16 / 8);    // 24 * 128 = 3072
        qkv_gemm_kernel<<<blocks, 256, stage_bytes, stream>>>(
            x, w_qkv, b_qkv, qarr, karr, varr);
    }
    // K2: one thread per (pixel, head)
    {
        int threads = NPIX * NH;                     // 65536
        attn_kernel<<<threads / 256, 256, 0, stream>>>(qarr, karr, varr, rpb, oarr);
    }
    // K3: 8 N-tiles x 128 M-blocks
    {
        int blocks = (CC / 16) * (NPIX / 16 / 8);    // 8 * 128 = 1024
        proj_gemm_kernel<<<blocks, 256, stage_bytes, stream>>>(
            oarr, w_proj, b_proj, out);
    }
}
